// RateRNN_78417512891124
// MI455X (gfx1250) — compile-verified, matrix operand order checked by reading, not run on hardware
//
#include <hip/hip_runtime.h>
#include <hip/hip_bf16.h>

// ---------------------------------------------------------------------------
// RateRNN on MI455X (gfx1250).
//
// The T=4096 scan is serial; per-step cost is bounded by streaming w_rec into
// the WGPs from L2 plus the dependent v_wmma chain. Design:
//   * w_rec/w_in converted to f16 and pre-packed into the per-lane VGPR
//     fragment layout of V_WMMA_F32_16X16X32_F16, so the hot loop is pure
//     coalesced global_load_b128 (saddr + loop-invariant 32-bit voffset +
//     immediate) + v_wmma, fp32 accumulation.
//   * hidden state lives in fp32 registers (each wave permanently owns two
//     16-wide column tiles); an f16 copy is shared through LDS each step.
//   * two accumulator chains per tile (even/odd K-blocks) halve the
//     dependent-WMMA latency chain; merged once per step.
//   * w_in (128 KB f16 fragments) cached in LDS for the whole kernel
//     (CDNA5 WGP LDS = 320 KB); readout (shfl_xor + ds_add_f32) fused.
//   * raw input tiles software-pipelined 2 steps ahead in registers, with
//     global_prefetch_b8 issued 8 steps ahead to cover HBM latency.
// 8 workgroups x 512 threads (16 waves); wave w owns N-tiles 2w, 2w+1.
// ---------------------------------------------------------------------------

typedef _Float16 v16h __attribute__((ext_vector_type(16)));
typedef float    v8f  __attribute__((ext_vector_type(8)));

#define T_ 4096
#define I_ 128
#define H_ 512
#define B_ 128
#define HS 536   // hbuf row stride in halves (512 + 24 pad: 16B aligned, bank-spread)
#define XS 152   // xbuf row stride in halves (128 + 24 pad)

// LDS carve (bytes): win_pk 131072 | hbuf 17152 | xbuf 9728 | yacc 128
#define LDS_WIN   0
#define LDS_HBUF  131072
#define LDS_XBUF  148224
#define LDS_YACC  157952
#define LDS_TOTAL 158080

union Frag  { v16h v; float4 f[2]; };
union Pack8 { _Float16 h[8]; float4 f; };
union Pack4 { _Float16 h[4]; uint2 u; };

// --- Pack w_rec (512x512 f32, row-major [g][k]) into f16 B-fragment layout ---
// Fragment (nt, kb) covers N = nt*16..+15, K = kb*32..+31.
// dst float4 index: (nt*16 + kb)*64 + j2*32 + lane   (j2 = K-halfword group)
__global__ __launch_bounds__(256) void pack_wrec_kernel(
    const float* __restrict__ wrec, float4* __restrict__ dst) {
  int c  = blockIdx.x * 256 + threadIdx.x;   // 32768 chunks of 8 halves
  int lc = c & 31;
  int j2 = (c >> 5) & 1;
  int kb = (c >> 6) & 15;
  int nt = c >> 10;
  int K  = kb * 32 + ((lc & 16) ? 8 : 0) + j2 * 16;
  int g  = nt * 16 + (lc & 15);
  const float* s = wrec + g * H_ + K;
  Pack8 u;
#pragma unroll
  for (int e = 0; e < 8; ++e) u.h[e] = (_Float16)s[e];
  dst[(unsigned)((nt * 16 + kb) * 64 + j2 * 32 + lc)] = u.f;
}

__device__ __forceinline__ float4 ldg16(const void* base, unsigned byteoff) {
  return *(const float4*)((const char*)base + byteoff);
}

// --- Persistent scan kernel: 8 blocks x 512 threads ---
__global__ __launch_bounds__(512, 1) void rate_rnn_scan(
    const float* __restrict__ inputs, const float* __restrict__ hidden,
    const float* __restrict__ w_in,   const float* __restrict__ b_rec,
    const float* __restrict__ w_out,  const float* __restrict__ b_out,
    const float4* __restrict__ wrec_pk, float* __restrict__ out) {
  extern __shared__ char smem[];
  float4*   win_pk = (float4*)(smem + LDS_WIN);     // w_in f16 fragments
  _Float16* hbuf   = (_Float16*)(smem + LDS_HBUF);  // 16 x HS
  _Float16* xbuf   = (_Float16*)(smem + LDS_XBUF);  // 2 x 16 x XS
  float*    yacc   = (float*)(smem + LDS_YACC);     // 2 x 16

  const int tid = threadIdx.x;
  const int l   = tid & 31;
  const int w   = tid >> 5;           // wave 0..15
  const int b0  = blockIdx.x * 16;    // batch tile base
  const int nt0 = 2 * w, nt1 = 2 * w + 1;
  const int lm  = l & 15;             // N within tile (B/C/D) and M (A)
  const int hi  = (l >> 4) & 1;       // lane half

  // ---- pack w_in (512x128 f32) into LDS f16 fragments (once) ----
  for (int c = tid; c < 8192; c += 512) {
    int lc = c & 31;
    int j2 = (c >> 5) & 1;
    int kb = (c >> 6) & 3;
    int nt = c >> 8;
    int K  = kb * 32 + ((lc & 16) ? 8 : 0) + j2 * 16;
    int g  = nt * 16 + (lc & 15);
    const float* s = w_in + g * I_ + K;
    Pack8 u;
#pragma unroll
    for (int e = 0; e < 8; ++e) u.h[e] = (_Float16)s[e];
    win_pk[c] = u.f;
  }
  if (tid < 32) yacc[tid] = 0.0f;

  // ---- per-lane constants ----
  const float brec0 = b_rec[nt0 * 16 + lm];
  const float brec1 = b_rec[nt1 * 16 + lm];
  const float wv0   = w_out[nt0 * 16 + lm];
  const float wv1   = w_out[nt1 * 16 + lm];
  const float bout  = b_out[0];

  // Loop-invariant per-lane byte bases; kb contribution becomes an
  // instruction-immediate offset (kb*1024 [+512]) in the hot loop.
  const unsigned wrb0 = (unsigned)((nt0 * 16 * 64 + l) * 16);  // w_rec frags
  const unsigned wrb1 = (unsigned)((nt1 * 16 * 64 + l) * 16);
  const unsigned wib0 = (unsigned)((nt0 * 4 * 64 + l) * 16);   // w_in frags
  const unsigned wib1 = (unsigned)((nt1 * 4 * 64 + l) * 16);

  // ---- init hidden state: regs (fp32) + LDS (f16) ----
  v8f h0, h1;
#pragma unroll
  for (int v = 0; v < 8; ++v) {
    int m = v + 8 * hi;
    h0[v] = hidden[(size_t)(b0 + m) * H_ + nt0 * 16 + lm];
    h1[v] = hidden[(size_t)(b0 + m) * H_ + nt1 * 16 + lm];
    hbuf[m * HS + nt0 * 16 + lm] = (_Float16)h0[v];
    hbuf[m * HS + nt1 * 16 + lm] = (_Float16)h1[v];
  }

  // ---- x pipeline: thread = (row, float4 column) of the 16x128 tile ----
  const int xr_row = w;        // 0..15
  const int xr_c4  = l;        // 0..31
  const float4* xg = (const float4*)inputs;
  const size_t xrowbase = (size_t)(b0 + xr_row) * T_ * 32 + xr_c4;
  {
    float4 x0 = xg[xrowbase];                 // x(0)
    Pack4 p;
    p.h[0] = (_Float16)x0.x; p.h[1] = (_Float16)x0.y;
    p.h[2] = (_Float16)x0.z; p.h[3] = (_Float16)x0.w;
    *(uint2*)&xbuf[xr_row * XS + xr_c4 * 4] = p.u;
  }
  float4 xr = xg[xrowbase + 32];              // x(1) in flight
  __syncthreads();

  const float4* hb4 = (const float4*)hbuf;

  for (int t = 0; t < T_; ++t) {
    const int p = t & 1;
    const float4* xb4 = (const float4*)(xbuf + p * 16 * XS);

    // Two accumulator chains per tile (even/odd kb) -> 4 independent
    // dependent-WMMA streams per wave instead of 2.
    v8f a0e, a0o, a1e, a1o;
#pragma unroll
    for (int v = 0; v < 8; ++v) {
      a0e[v] = brec0; a1e[v] = brec1;
      a0o[v] = 0.0f;  a1o[v] = 0.0f;
    }

    // ---- input projection: acc += x_t @ w_in^T  (K = 128) ----
#pragma unroll
    for (int kb = 0; kb < 4; ++kb) {
      Frag A, Bm0, Bm1;
      int ai = lm * (XS / 8) + kb * 4 + hi;
      A.f[0] = xb4[ai];
      A.f[1] = xb4[ai + 2];
      Bm0.f[0] = ldg16(win_pk, wib0 + (unsigned)(kb * 1024));
      Bm0.f[1] = ldg16(win_pk, wib0 + (unsigned)(kb * 1024 + 512));
      Bm1.f[0] = ldg16(win_pk, wib1 + (unsigned)(kb * 1024));
      Bm1.f[1] = ldg16(win_pk, wib1 + (unsigned)(kb * 1024 + 512));
      if (kb & 1) {
        a0o = __builtin_amdgcn_wmma_f32_16x16x32_f16(false, A.v, false, Bm0.v,
                                                     (short)0, a0o, false, false);
        a1o = __builtin_amdgcn_wmma_f32_16x16x32_f16(false, A.v, false, Bm1.v,
                                                     (short)0, a1o, false, false);
      } else {
        a0e = __builtin_amdgcn_wmma_f32_16x16x32_f16(false, A.v, false, Bm0.v,
                                                     (short)0, a0e, false, false);
        a1e = __builtin_amdgcn_wmma_f32_16x16x32_f16(false, A.v, false, Bm1.v,
                                                     (short)0, a1e, false, false);
      }
    }

    // ---- recurrence: acc += h @ w_rec^T  (K = 512), w_rec streamed from L2;
    //      address = SGPR base + loop-invariant voffset + immediate ----
#pragma unroll 4
    for (int kb = 0; kb < 16; ++kb) {
      Frag A, Bm0, Bm1;
      int ai = lm * (HS / 8) + kb * 4 + hi;
      A.f[0] = hb4[ai];
      A.f[1] = hb4[ai + 2];
      Bm0.f[0] = ldg16(wrec_pk, wrb0 + (unsigned)(kb * 1024));
      Bm0.f[1] = ldg16(wrec_pk, wrb0 + (unsigned)(kb * 1024 + 512));
      Bm1.f[0] = ldg16(wrec_pk, wrb1 + (unsigned)(kb * 1024));
      Bm1.f[1] = ldg16(wrec_pk, wrb1 + (unsigned)(kb * 1024 + 512));
      if (kb & 1) {
        a0o = __builtin_amdgcn_wmma_f32_16x16x32_f16(false, A.v, false, Bm0.v,
                                                     (short)0, a0o, false, false);
        a1o = __builtin_amdgcn_wmma_f32_16x16x32_f16(false, A.v, false, Bm1.v,
                                                     (short)0, a1o, false, false);
      } else {
        a0e = __builtin_amdgcn_wmma_f32_16x16x32_f16(false, A.v, false, Bm0.v,
                                                     (short)0, a0e, false, false);
        a1e = __builtin_amdgcn_wmma_f32_16x16x32_f16(false, A.v, false, Bm1.v,
                                                     (short)0, a1e, false, false);
      }
    }

    // ---- h_new = 0.8*h + 0.2*relu(acc)  (fp32, in registers) ----
#pragma unroll
    for (int v = 0; v < 8; ++v) {
      float s0 = a0e[v] + a0o[v];
      float s1 = a1e[v] + a1o[v];
      float r0 = s0 > 0.0f ? s0 : 0.0f;
      float r1 = s1 > 0.0f ? s1 : 0.0f;
      h0[v] = 0.8f * h0[v] + 0.2f * r0;
      h1[v] = 0.8f * h1[v] + 0.2f * r1;
    }

    __syncthreads();  // all reads of hbuf / xbuf[p] complete

    // ---- publish h_new (f16) for next step's A fragments ----
#pragma unroll
    for (int v = 0; v < 8; ++v) {
      int m = v + 8 * hi;
      hbuf[m * HS + nt0 * 16 + lm] = (_Float16)h0[v];
      hbuf[m * HS + nt1 * 16 + lm] = (_Float16)h1[v];
    }
    // ---- stage x(t+1) into the other buffer; launch x(t+2) load ----
    {
      Pack4 px;
      px.h[0] = (_Float16)xr.x; px.h[1] = (_Float16)xr.y;
      px.h[2] = (_Float16)xr.z; px.h[3] = (_Float16)xr.w;
      *(uint2*)&xbuf[(p ^ 1) * 16 * XS + xr_row * XS + xr_c4 * 4] = px.u;
    }
    {
      int tn = t + 2 < T_ ? t + 2 : T_ - 1;
      xr = xg[xrowbase + (size_t)tn * 32];
      // Warm the cache for the input stream well ahead of the register
      // pipeline (emits global_prefetch_b8 on gfx1250).
      int tp = t + 8 < T_ ? t + 8 : T_ - 1;
      __builtin_prefetch((const void*)(xg + xrowbase + (size_t)tp * 32), 0, 1);
    }

    // ---- fused readout: y[b,t] = w_out . h_new + b_out ----
#pragma unroll
    for (int v = 0; v < 8; ++v) {
      float s = h0[v] * wv0 + h1[v] * wv1;
      s += __shfl_xor(s, 1);
      s += __shfl_xor(s, 2);
      s += __shfl_xor(s, 4);
      s += __shfl_xor(s, 8);
      if (lm == 0) atomicAdd(&yacc[p * 16 + v + 8 * hi], s);
    }

    __syncthreads();  // h_new stores + y partials complete

    if (w == 15 && l < 16) {
      out[(size_t)(b0 + l) * T_ + t] = yacc[p * 16 + l] + bout;
      yacc[p * 16 + l] = 0.0f;      // recycle for step t+2
    }
  }

  // ---- h_final (fp32, straight from registers) ----
  float* hf = out + (size_t)B_ * T_;
#pragma unroll
  for (int v = 0; v < 8; ++v) {
    int m = v + 8 * hi;
    hf[(size_t)(b0 + m) * H_ + nt0 * 16 + lm] = h0[v];
    hf[(size_t)(b0 + m) * H_ + nt1 * 16 + lm] = h1[v];
  }
}

extern "C" void kernel_launch(void* const* d_in, const int* in_sizes, int n_in,
                              void* d_out, int out_size, void* d_ws, size_t ws_size,
                              hipStream_t stream) {
  const float* inputs = (const float*)d_in[0];
  const float* hidden = (const float*)d_in[1];
  const float* w_in   = (const float*)d_in[2];
  const float* w_rec  = (const float*)d_in[3];
  const float* b_rec  = (const float*)d_in[4];
  const float* w_out  = (const float*)d_in[5];
  const float* b_out  = (const float*)d_in[6];
  float*  out     = (float*)d_out;
  float4* wrec_pk = (float4*)d_ws;   // 512 KB of scratch for f16-packed w_rec

  (void)in_sizes; (void)n_in; (void)out_size; (void)ws_size;

  // Allow >64KB dynamic LDS (CDNA5 WGP supports up to 320KB per workgroup).
  (void)hipFuncSetAttribute((const void*)rate_rnn_scan,
                            hipFuncAttributeMaxDynamicSharedMemorySize,
                            LDS_TOTAL);

  pack_wrec_kernel<<<128, 256, 0, stream>>>(w_rec, wrec_pk);
  rate_rnn_scan<<<8, 512, LDS_TOTAL, stream>>>(inputs, hidden, w_in, b_rec,
                                               w_out, b_out, wrec_pk, out);
}
